// DeformVoxResBlock3d_28149215658690
// MI455X (gfx1250) — matrix operand order repaired and checked by reading, not compiled, not run on hardware
//
#include <hip/hip_runtime.h>
#include <hip/hip_bf16.h>

// ---------------------------------------------------------------------------
// DeformVoxResBlock3d on gfx1250 (MI455X): both GEMMs on v_wmma_f32_16x16x32_f16
// B=2, C=16, D=H=W=48, taps=27, GEMM-K = tap*16 + channel, 432 padded to 448.
// Round 3: tap-major K ordering + channel-fastest halo tile -> every B fragment
// (both GEMMs) is two 16B LDS vector loads; sampling stores are vectorized.
// Goal: VGPR usage back under 256 (no s_set_vgpr_msb), minimal VALU overhead.
// ---------------------------------------------------------------------------

typedef __attribute__((ext_vector_type(16))) _Float16 v16h;
typedef __attribute__((ext_vector_type(8)))  _Float16 v8h;
typedef __attribute__((ext_vector_type(8)))  float    v8f;

#define DIM      48
#define DD       110592          // 48^3
#define NCH      16
#define KTAPS    27
#define KDENSE   432             // 27 * 16
#define KPAD     448             // padded to multiple of 32
#define MOFF     96              // 81 offset channels padded to 96
#define NVOX     221184          // B * 48^3
#define NOUT     3538944         // B * C * 48^3
#define CNT_INV  (1.0f / 221184.0f)
#define BMLD     456             // bm row stride in elements (bank-spread pad)

__device__ __forceinline__ v16h cat8(v8h lo, v8h hi) {
  return __builtin_shufflevector(lo, hi, 0, 1, 2, 3, 4, 5, 6, 7,
                                 8, 9, 10, 11, 12, 13, 14, 15);
}

// ---- A fragment (16-bit A 16x32): lane m = lane&15.
// lanes 0-15 hold K 0..15; lanes 16-31 hold K 16..31.
// -> two aligned 16B global vector loads per fragment.
__device__ __forceinline__ v16h load_a_frag(const _Float16* __restrict__ W,
                                            int row, int kbase, int khalf8) {
  const _Float16* p = W + row * KPAD + kbase + khalf8;
  return cat8(*(const v8h*)p, *(const v8h*)(p + 8));
}

// ---- B fragment (16-bit B 32x16) from column-major LDS bm2[n][456].
// lane n = lane&15; lanes 0-15 hold K 0..15, lanes 16-31 hold K 16..31
// -> 16 consecutive f16 = two aligned 16B LDS vector loads.
__device__ __forceinline__ v16h load_b_frag(const _Float16* bmRow,
                                            int kbase, int khalf16) {
  const _Float16* p = bmRow + kbase + khalf16;
  return cat8(*(const v8h*)p, *(const v8h*)(p + 8));
}

// ---- B fragment from the channel-fastest halo tile (virtual im2col).
// K = t*16 + c: a 32-wide K-step covers taps {t0, t0+1}; a lane's 16 elements
// are the 16 channels at one tap -> contiguous in xt -> two ds_load_b128.
__device__ __forceinline__ v16h make_b_frag_x(
    const _Float16 (&xt)[3][3][18][NCH], int kb, bool hiLane, int n) {
  const int t0 = kb >> 4;                     // constant after full unroll
  const int tz0 = t0 / 9, ty0 = (t0 / 3) % 3, tx0 = t0 % 3;
  const _Float16* p0 = &xt[tz0][ty0][tx0 + n][0];
  const int t1 = t0 + 1;
  if (t1 < KTAPS) {
    const int tz1 = t1 / 9, ty1 = (t1 / 3) % 3, tx1 = t1 % 3;
    const _Float16* p1 = &xt[tz1][ty1][tx1 + n][0];
    const _Float16* p = hiLane ? p1 : p0;     // one select per fragment
    return cat8(*(const v8h*)p, *(const v8h*)(p + 8));
  } else {                                    // last K-step: hi half is padding
    const v16h b = cat8(*(const v8h*)p0, *(const v8h*)(p0 + 8));
    const v16h z = {};
    return hiLane ? z : b;
  }
}

// ---------------------------------------------------------------------------
// Weight repack: fp32 conv weights -> padded f16 GEMM matrices, K = t*16 + c
// ---------------------------------------------------------------------------
__global__ void pack_weights(const float* __restrict__ w_off,
                             const float* __restrict__ w_main,
                             _Float16* __restrict__ Woff,
                             _Float16* __restrict__ Wmain) {
  const int tid = blockIdx.x * blockDim.x + threadIdx.x;
  const int stride = gridDim.x * blockDim.x;
  for (int i = tid; i < MOFF * KPAD; i += stride) {
    const int r = i / KPAD, k = i % KPAD;
    const int t = k >> 4, c = k & 15;
    float v = 0.0f;
    if (r < 81 && t < KTAPS) v = w_off[(r * NCH + c) * KTAPS + t];
    Woff[i] = (_Float16)v;
  }
  for (int i = tid; i < NCH * KPAD; i += stride) {
    const int r = i / KPAD, k = i % KPAD;
    const int t = k >> 4, c = k & 15;
    float v = 0.0f;
    if (t < KTAPS) v = w_main[(r * NCH + c) * KTAPS + t];
    Wmain[i] = (_Float16)v;
  }
}

// ---------------------------------------------------------------------------
// Fused: offset conv (WMMA) -> trilinear deformable sampling -> main conv
// (WMMA) -> h + batch statistics. One wave per 16-voxel tile along W.
// ---------------------------------------------------------------------------
__global__ __launch_bounds__(32) void deform_fused(
    const float* __restrict__ x, const float* __restrict__ b_off,
    const _Float16* __restrict__ Woff, const _Float16* __restrict__ Wmain,
    float* __restrict__ hbuf, float* __restrict__ stats) {
  __shared__ _Float16 xt[3][3][18][NCH];   // halo tile, channel-fastest
  __shared__ float    offL[81][16];        // predicted offsets for this tile
  __shared__ _Float16 bm2[16][BMLD];       // sampled columns, [voxel][K]
  __shared__ float    hT[16][16];          // h tile for stats reduction

  const int lane = threadIdx.x;
  const int tile = blockIdx.x;
  const int x0 = (tile % 3) * 16;
  const int y  = (tile / 3) % DIM;
  const int z  = (tile / 144) % DIM;
  const int b  = tile / 6912;
  const float* xb = x + b * NCH * DD;

  const int  n       = lane & 15;
  const bool hiLane  = (lane >= 16);
  const int  khalf8  = hiLane ? 16 : 0;   // A-fragment K-half offset
  const int  khalf16 = hiLane ? 16 : 0;   // B-fragment K-half offset

  // ---- stage halo x-tile into LDS (channel-fastest, zero outside volume) ----
  for (int i = lane; i < NCH * 3 * 3 * 18; i += 32) {   // 2592 = 81*32, uniform
    const int c = i / 162, r = i % 162;                 // xx fastest: coalesced
    const int zz = r / 54, yy = (r / 18) % 3, xx = r % 18;
    const int gz = z + zz - 1, gy = y + yy - 1, gx = x0 + xx - 1;
    float v = 0.0f;
    if (gz >= 0 && gz < DIM && gy >= 0 && gy < DIM && gx >= 0 && gx < DIM)
      v = xb[c * DD + (gz * DIM + gy) * DIM + gx];
    xt[zz][yy][xx][c] = (_Float16)v;
  }
  __syncthreads();

  // ---- offset conv GEMM: [96 x 448] * [448 x 16], software-pipelined A ----
  v8f acc[6] = {};
  {
    v16h af[6];
#pragma unroll
    for (int mt = 0; mt < 6; ++mt)
      af[mt] = load_a_frag(Woff, mt * 16 + n, 0, khalf8);

#pragma unroll
    for (int kb = 0; kb < KPAD; kb += 32) {
      const v16h bf = make_b_frag_x(xt, kb, hiLane, n);
      v16h afn[6];
      if (kb + 32 < KPAD) {
#pragma unroll
        for (int mt = 0; mt < 6; ++mt)
          afn[mt] = load_a_frag(Woff, mt * 16 + n, kb + 32, khalf8);
      }
#pragma unroll
      for (int mt = 0; mt < 6; ++mt)
        acc[mt] = __builtin_amdgcn_wmma_f32_16x16x32_f16(
            false, af[mt], false, bf, (short)0, acc[mt], false, false);
      if (kb + 32 < KPAD) {
#pragma unroll
        for (int mt = 0; mt < 6; ++mt) af[mt] = afn[mt];
      }
    }
  }
  {
    const int mo = hiLane ? 8 : 0;
#pragma unroll
    for (int mt = 0; mt < 6; ++mt)
#pragma unroll
      for (int v = 0; v < 8; ++v) {
        const int row = mt * 16 + v + mo;
        if (row < 81) offL[row][n] = acc[mt][v] + b_off[row];
      }
  }
  __syncthreads();

  // ---- deformable trilinear sampling -> bm2[n][K], K = t*16 + c ----
  for (int i = lane; i < 16 * (BMLD - KDENSE); i += 32) {  // zero K-padding
    const int col = i / (BMLD - KDENSE);
    const int kk  = KDENSE + i % (BMLD - KDENSE);
    bm2[col][kk] = (_Float16)0.0f;
  }
  for (int idx = lane; idx < KTAPS * 16; idx += 32) {      // 432 (tap, voxel) pairs
    const int t = idx >> 4, nn = idx & 15;
    const int tz = t / 9, ty = (t / 3) % 3, tx = t % 3;
    const float pz = (float)(z + tz - 1)       + offL[t * 3 + 0][nn];
    const float py = (float)(y + ty - 1)       + offL[t * 3 + 1][nn];
    const float px = (float)(x0 + nn + tx - 1) + offL[t * 3 + 2][nn];
    const float zf = floorf(pz), yf = floorf(py), xf = floorf(px);
    const int iz = (int)zf, iy = (int)yf, ix = (int)xf;
    const float fz = pz - zf, fy = py - yf, fx = px - xf;
    float s[NCH] = {};
#pragma unroll
    for (int cr = 0; cr < 8; ++cr) {
      const int cz = iz + (cr >> 2), cy = iy + ((cr >> 1) & 1), cx = ix + (cr & 1);
      const float w = ((cr & 4) ? fz : 1.0f - fz) *
                      ((cr & 2) ? fy : 1.0f - fy) *
                      ((cr & 1) ? fx : 1.0f - fx);
      if (cz >= 0 && cz < DIM && cy >= 0 && cy < DIM && cx >= 0 && cx < DIM) {
        const float* xp = xb + (cz * DIM + cy) * DIM + cx;
#pragma unroll
        for (int c = 0; c < NCH; ++c) s[c] += w * xp[c * DD];
      }
    }
    v8h lo, hi;                                  // 16 channels contiguous in K
#pragma unroll
    for (int c = 0; c < 8; ++c) {
      lo[c] = (_Float16)s[c];
      hi[c] = (_Float16)s[c + 8];
    }
    *(v8h*)&bm2[nn][t * 16]     = lo;            // two 16B LDS vector stores
    *(v8h*)&bm2[nn][t * 16 + 8] = hi;
  }
  __syncthreads();

  // ---- main conv GEMM: [16 x 448] * [448 x 16], software-pipelined A ----
  v8f hacc = {};
  {
    const _Float16* bmRow = &bm2[n][0];
    v16h af = load_a_frag(Wmain, n, 0, khalf8);
#pragma unroll
    for (int kb = 0; kb < KPAD; kb += 32) {
      const v16h bf = load_b_frag(bmRow, kb, khalf16);
      v16h afn;
      if (kb + 32 < KPAD) afn = load_a_frag(Wmain, n, kb + 32, khalf8);
      hacc = __builtin_amdgcn_wmma_f32_16x16x32_f16(
          false, af, false, bf, (short)0, hacc, false, false);
      if (kb + 32 < KPAD) af = afn;
    }
  }

  // ---- emit h and per-channel batch statistics ----
  {
    const int mo = hiLane ? 8 : 0;
    float* hb = hbuf + b * NCH * DD + (z * DIM + y) * DIM + x0;
#pragma unroll
    for (int v = 0; v < 8; ++v) {
      const int m = v + mo;
      const float val = hacc[v];
      hT[m][n] = val;
      hb[m * DD + n] = val;
    }
  }
  __syncthreads();
  if (lane < NCH) {
    float s = 0.0f, s2 = 0.0f;
#pragma unroll
    for (int c = 0; c < 16; ++c) { const float v = hT[lane][c]; s += v; s2 += v * v; }
    atomicAdd(&stats[lane], s);
    atomicAdd(&stats[NCH + lane], s2);
  }
}

// ---------------------------------------------------------------------------
// BatchNorm (batch stats) + ReLU + residual add, 4 elements / thread
// ---------------------------------------------------------------------------
__global__ void finalize(const float* __restrict__ x, const float* __restrict__ h,
                         const float* __restrict__ stats,
                         const float* __restrict__ gamma, const float* __restrict__ beta,
                         float* __restrict__ out) {
  __shared__ float sa[NCH], sb[NCH];
  if (threadIdx.x < NCH) {
    const int c = threadIdx.x;
    const float mean = stats[c] * CNT_INV;
    const float var  = stats[NCH + c] * CNT_INV - mean * mean;
    const float g = gamma[c] * rsqrtf(var + 1e-5f);
    sa[c] = g;
    sb[c] = beta[c] - mean * g;
  }
  __syncthreads();
  const int i4 = (blockIdx.x * blockDim.x + threadIdx.x) * 4;
  if (i4 < NOUT) {
    const int c = (i4 / DD) & 15;         // DD % 4 == 0 -> same channel for all 4
    const float4 hv = *(const float4*)(h + i4);
    const float4 xv = *(const float4*)(x + i4);
    float4 o;
    o.x = xv.x + fmaxf(hv.x * sa[c] + sb[c], 0.0f);
    o.y = xv.y + fmaxf(hv.y * sa[c] + sb[c], 0.0f);
    o.z = xv.z + fmaxf(hv.z * sa[c] + sb[c], 0.0f);
    o.w = xv.w + fmaxf(hv.w * sa[c] + sb[c], 0.0f);
    *(float4*)(out + i4) = o;
  }
}

// ---------------------------------------------------------------------------
extern "C" void kernel_launch(void* const* d_in, const int* in_sizes, int n_in,
                              void* d_out, int out_size, void* d_ws, size_t ws_size,
                              hipStream_t stream) {
  const float* x      = (const float*)d_in[0];
  const float* w_off  = (const float*)d_in[1];
  const float* b_off  = (const float*)d_in[2];
  const float* w_main = (const float*)d_in[3];
  const float* gamma  = (const float*)d_in[4];
  const float* beta   = (const float*)d_in[5];
  float* out = (float*)d_out;

  char* ws = (char*)d_ws;
  _Float16* Woff  = (_Float16*)(ws);                       // 96*448*2   = 86016 B
  _Float16* Wmain = (_Float16*)(ws + 86016);               // 16*448*2   = 14336 B
  float*    stats = (float*)(ws + 100352);                 // 32 floats  =   128 B
  float*    hbuf  = (float*)(ws + 100608);                 // 2*16*48^3*4 ≈ 14.2 MB

  hipMemsetAsync(stats, 0, 32 * sizeof(float), stream);
  pack_weights<<<32, 256, 0, stream>>>(w_off, w_main, Woff, Wmain);
  deform_fused<<<NVOX / 16, 32, 0, stream>>>(x, b_off, Woff, Wmain, hbuf, stats);
  finalize<<<NOUT / 1024, 256, 0, stream>>>(x, hbuf, stats, gamma, beta, out);
}
